// MSARowAttentionV44_29197187678430
// MI455X (gfx1250) — matrix operand order repaired and checked by reading, not compiled, not run on hardware
//
#include <hip/hip_runtime.h>
#include <hip/hip_bf16.h>
#include <stdint.h>

#define S_   128
#define N_   256
#define DIM_ 256
#define H_   8
#define DH_  32
#define PD_  128
#define SCALE_ 0.17677669529663687f   // 32^-0.5

typedef __attribute__((ext_vector_type(16))) __bf16 v16bf;
typedef __attribute__((ext_vector_type(8)))  __bf16 v8bf;
typedef __attribute__((ext_vector_type(8)))  float  v8f;

// ---- WMMA fragment helpers (layouts per CDNA5 ISA 7.12.2, wave32) ----
// A 16x32 bf16: lane L holds row M=L&15; K = (L<16 ? 0 : 8) + {0..7, 16..23}
__device__ __forceinline__ v16bf frag_a(const __bf16* p, int stride) {
  int lane = threadIdx.x & 31;
  const __bf16* q = p + (lane & 15) * stride + ((lane >> 4) << 3);
  v8bf lo = *(const v8bf*)(q);
  v8bf hi = *(const v8bf*)(q + 16);
  v16bf o;
#pragma unroll
  for (int i = 0; i < 8; ++i) { o[i] = lo[i]; o[i + 8] = hi[i]; }
  return o;
}
// B 32x16 bf16: lane L holds col N=L&15; K = (L<16 ? 0 : 8) + {0..7, 16..23}
__device__ __forceinline__ v16bf frag_b(const __bf16* p, int stride) {
  int lane = threadIdx.x & 31;
  const __bf16* q = p + (lane & 15) * stride + ((lane >> 4) << 3);
  v8bf lo = *(const v8bf*)(q);
  v8bf hi = *(const v8bf*)(q + 16);
  v16bf o;
#pragma unroll
  for (int i = 0; i < 8; ++i) { o[i] = lo[i]; o[i + 8] = hi[i]; }
  return o;
}
// C/D 16x16 f32: lane L col N=L&15; VGPR r -> row M = r + (L<16 ? 0 : 8)
__device__ __forceinline__ void frag_st(float* p, int stride, v8f a) {
  int lane = threadIdx.x & 31;
  float* q = p + (lane & 15) + (((lane >> 4) << 3) * stride);
#pragma unroll
  for (int r = 0; r < 8; ++r) q[r * stride] = a[r];
}
__device__ __forceinline__ v8f wmma_bf16(v16bf a, v16bf b, v8f c) {
  return __builtin_amdgcn_wmma_f32_16x16x32_bf16(false, a, false, b,
                                                 (short)0, c, false, false);
}

// ---------------- Kernel 1: QKV projection (bf16 WMMA GEMM) ----------------
// Y[m, c] = msa[m,:] . Wqkv[c,:] + bqkv[c]; scatter to head-major bf16 bufs.
__global__ __launch_bounds__(256) void k_qkv(const float* __restrict__ msa,
    const float* __restrict__ Wqkv, const float* __restrict__ bqkv,
    __bf16* __restrict__ qbf, __bf16* __restrict__ kbf, __bf16* __restrict__ vbf)
{
  __shared__ __align__(16) __bf16 Al[64 * 32];
  __shared__ __align__(16) __bf16 Bl[64 * 32];
  const int tid = threadIdx.x;
  const int wave = tid >> 5;
  const int lane = tid & 31;
  const int m0 = blockIdx.x * 64;    // rows (s*256+n)
  const int c0 = blockIdx.y * 64;    // cols of 768
  const int wm = (wave >> 2) * 32;
  const int wn = (wave & 3) * 16;
  v8f acc0 = {}; v8f acc1 = {};
  for (int kb = 0; kb < DIM_; kb += 32) {
    __syncthreads();
#pragma unroll
    for (int t = 0; t < 8; ++t) {
      int idx = t * 256 + tid;
      int r = idx >> 5, k = idx & 31;
      Al[idx] = (__bf16)msa[(size_t)(m0 + r) * DIM_ + kb + k];
      Bl[idx] = (__bf16)Wqkv[(size_t)(c0 + r) * DIM_ + kb + k];
    }
    __syncthreads();
    v16bf b  = frag_b(&Bl[wn * 32], 32);
    v16bf a0 = frag_a(&Al[(wm + 0) * 32], 32);
    v16bf a1 = frag_a(&Al[(wm + 16) * 32], 32);
    acc0 = wmma_bf16(a0, b, acc0);
    acc1 = wmma_bf16(a1, b, acc1);
  }
  const int c = c0 + wn + (lane & 15);
  const int rb = (lane >> 4) << 3;
  const float bias = bqkv[c];
  const int which = c >> 8;
  const int h = (c >> 5) & 7;
  const int dh = c & 31;
#pragma unroll
  for (int t = 0; t < 2; ++t) {
    v8f acc = t ? acc1 : acc0;
#pragma unroll
    for (int r = 0; r < 8; ++r) {
      int m = m0 + wm + t * 16 + rb + r;
      int s = m >> 8, n = m & 255;
      float v = acc[r] + bias;
      if (which == 0)
        qbf[((size_t)(s * H_ + h) * N_ + n) * DH_ + dh] = (__bf16)(v * SCALE_);
      else if (which == 1)
        kbf[((size_t)(s * H_ + h) * N_ + n) * DH_ + dh] = (__bf16)v;
      else  // V stored transposed [s][h][dh][n] for B-fragment loads in P@V
        vbf[((size_t)(s * H_ + h) * DH_ + dh) * N_ + n] = (__bf16)v;
    }
  }
}

// ---------------- Kernel 2: pair bias (async-LDS staged GEMV) ----------------
// pb[h][i*256+j] = pair[i,j,:] . Wpb[h,:] + bpb[h]
__global__ __launch_bounds__(256) void k_pairbias(const float* __restrict__ pair,
    const float* __restrict__ Wpb, const float* __restrict__ bpb,
    float* __restrict__ pb)
{
  __shared__ __align__(16) float rows[32 * 128];
  __shared__ float wlds[8 * 128];
  const int tid = threadIdx.x;
  const int rowbase = blockIdx.x * 32;
#pragma unroll
  for (int t = 0; t < 4; ++t) {
    int idx = t * 256 + tid;
    wlds[idx] = Wpb[idx];
  }
  // gfx1250 async global->LDS copy: 1024 x 16B chunks, tracked by ASYNCcnt
#pragma unroll
  for (int t = 0; t < 4; ++t) {
    int idx = t * 256 + tid;
    unsigned loff = (unsigned)(uintptr_t)(&rows[0]) + (unsigned)idx * 16u;
    const float* g = pair + (size_t)rowbase * PD_ + (size_t)idx * 4;
    asm volatile("global_load_async_to_lds_b128 %0, %1, off"
                 :: "v"(loff), "v"((unsigned long long)(uintptr_t)g)
                 : "memory");
  }
  asm volatile("s_wait_asynccnt 0" ::: "memory");
  __syncthreads();
  const int row = tid >> 3, h = tid & 7;
  const float* r = &rows[row * 128];
  const float* w = &wlds[h * 128];
  float sum = 0.f;
#pragma unroll 4
  for (int c2 = 0; c2 < 128; ++c2) sum += r[c2] * w[c2];
  pb[(size_t)h * (N_ * N_) + rowbase + row] = sum + bpb[h];
}

// ---------------- Kernel 3: attention (QK^T, softmax, P@V) ----------------
__global__ __launch_bounds__(256) void k_attn(const __bf16* __restrict__ qbf,
    const __bf16* __restrict__ kbf, const __bf16* __restrict__ vbf,
    const float* __restrict__ pb, const int* __restrict__ mask,
    __bf16* __restrict__ obf)
{
  extern __shared__ char smem[];
  float*  sc = (float*)smem;                    // 64x256 f32 scores (64KB)
  __bf16* Pb = (__bf16*)(smem + 64 * 256 * 4);  // 64x256 bf16 probs (32KB)
  const int tid = threadIdx.x, lane = tid & 31, wave = tid >> 5;
  const int bx = blockIdx.x;
  const int qt = bx & 3, h = (bx >> 2) & 7, s = bx >> 5;
  const int qbase = qt * 64;
  const __bf16* qh = qbf + ((size_t)s * H_ + h) * N_ * DH_;
  const __bf16* kh = kbf + ((size_t)s * H_ + h) * N_ * DH_;
  const __bf16* vh = vbf + ((size_t)s * H_ + h) * DH_ * N_;
  // phase 1: raw scores (q pre-scaled), K=DH=32 single WMMA per tile
  {
    const int mrow = (wave & 3) * 16;
    const int ch = (wave >> 2) * 128;
    v16bf a = frag_a(qh + (size_t)(qbase + mrow) * DH_, DH_);
#pragma unroll
    for (int j = 0; j < 8; ++j) {
      int col = ch + j * 16;
      v16bf b = frag_b(kh + (size_t)col * DH_, DH_);
      v8f acc = {};
      acc = wmma_bf16(a, b, acc);
      frag_st(sc + mrow * 256 + col, 256, acc);
    }
  }
  __syncthreads();
  // phase 1b: + pair bias, mask (coalesced)
  for (int idx = tid; idx < 64 * 256; idx += 256) {
    int i = idx >> 8, j = idx & 255;
    int gi = qbase + i;
    float v = sc[idx] + pb[(size_t)h * (N_ * N_) + gi * 256 + j];
    if (mask[gi * 256 + j] == 0) v = -1.0e9f;
    sc[idx] = v;
  }
  __syncthreads();
  // phase 2: softmax, 4 lanes per row (shuffle reduce within wave32)
  {
    const int row = tid >> 2, part = tid & 3;
    const float* r = sc + row * 256 + part * 64;
    float m = -3.4e38f;
    for (int j = 0; j < 64; ++j) m = fmaxf(m, r[j]);
    m = fmaxf(m, __shfl_xor(m, 1));
    m = fmaxf(m, __shfl_xor(m, 2));
    float sum = 0.f;
    for (int j = 0; j < 64; ++j) sum += __expf(r[j] - m);
    sum += __shfl_xor(sum, 1);
    sum += __shfl_xor(sum, 2);
    float inv = 1.f / sum;
    __bf16* pr = Pb + row * 256 + part * 64;
    for (int j = 0; j < 64; ++j) pr[j] = (__bf16)(__expf(r[j] - m) * inv);
  }
  __syncthreads();
  // phase 3: O = P(64x256) @ V(256x32), write in out-proj A layout
  {
    const int mt = (wave & 3) * 16;
    const int nh = (wave >> 2) * 16;
    v8f acc = {};
#pragma unroll
    for (int kb = 0; kb < N_; kb += 32) {
      v16bf a = frag_a(Pb + mt * 256 + kb, 256);
      v16bf b = frag_b(vh + (size_t)nh * N_ + kb, N_);
      acc = wmma_bf16(a, b, acc);
    }
    const int c = nh + (lane & 15);
    const int rb = (lane >> 4) << 3;
#pragma unroll
    for (int r = 0; r < 8; ++r) {
      int n = qbase + mt + rb + r;
      obf[((size_t)(s * N_ + n)) * DIM_ + h * DH_ + c] = (__bf16)acc[r];
    }
  }
}

// ------- Kernel 4: out projection + bias + residual + LayerNorm -------
__global__ __launch_bounds__(256) void k_outln(const __bf16* __restrict__ obf,
    const float* __restrict__ Wout, const float* __restrict__ bout,
    const float* __restrict__ msa, const float* __restrict__ gamma,
    const float* __restrict__ beta, float* __restrict__ out)
{
  __shared__ __align__(16) __bf16 Bl[256 * 32];
  __shared__ float xb[32 * 256];
  const int tid = threadIdx.x, lane = tid & 31, wave = tid >> 5;
  const int m0 = blockIdx.x * 32;
  const int mt = (wave & 1) * 16;
  const int nq = (wave >> 1) * 64;
  v8f acc[4] = {};
  for (int kb = 0; kb < DIM_; kb += 32) {
    __syncthreads();
#pragma unroll
    for (int t = 0; t < 32; ++t) {
      int idx = t * 256 + tid;
      int c = idx >> 5, k = idx & 31;
      Bl[idx] = (__bf16)Wout[(size_t)c * DIM_ + kb + k];
    }
    __syncthreads();
    v16bf a = frag_a(obf + (size_t)(m0 + mt) * DIM_ + kb, DIM_);
#pragma unroll
    for (int jn = 0; jn < 4; ++jn) {
      v16bf b = frag_b(&Bl[(nq + jn * 16) * 32], 32);
      acc[jn] = wmma_bf16(a, b, acc[jn]);
    }
  }
  const int rb = (lane >> 4) << 3;
#pragma unroll
  for (int jn = 0; jn < 4; ++jn) {
    int c = nq + jn * 16 + (lane & 15);
#pragma unroll
    for (int r = 0; r < 8; ++r) {
      int lr = mt + rb + r;
      int m = m0 + lr;
      xb[lr * 256 + c] = acc[jn][r] + bout[c] + msa[(size_t)m * DIM_ + c];
    }
  }
  __syncthreads();
  // LayerNorm: 8 lanes per row
  const int row = tid >> 3, part = tid & 7;
  const float* xr = xb + row * 256;
  float sum = 0.f, sq = 0.f;
  for (int j = part * 32; j < part * 32 + 32; ++j) {
    float v = xr[j]; sum += v; sq += v * v;
  }
  sum += __shfl_xor(sum, 1); sq += __shfl_xor(sq, 1);
  sum += __shfl_xor(sum, 2); sq += __shfl_xor(sq, 2);
  sum += __shfl_xor(sum, 4); sq += __shfl_xor(sq, 4);
  float mu = sum * (1.f / 256.f);
  float var = sq * (1.f / 256.f) - mu * mu;
  float rs = rsqrtf(var + 1e-5f);
  float* orow = out + (size_t)(m0 + row) * DIM_;
  for (int j = part * 32; j < part * 32 + 32; ++j)
    orow[j] = (xr[j] - mu) * rs * gamma[j] + beta[j];
}

extern "C" void kernel_launch(void* const* d_in, const int* in_sizes, int n_in,
                              void* d_out, int out_size, void* d_ws, size_t ws_size,
                              hipStream_t stream)
{
  const float* msa   = (const float*)d_in[0];
  const float* pair  = (const float*)d_in[1];
  const int*   mask  = (const int*)d_in[2];
  const float* Wqkv  = (const float*)d_in[3];
  const float* bqkv  = (const float*)d_in[4];
  const float* Wout  = (const float*)d_in[5];
  const float* bout  = (const float*)d_in[6];
  const float* Wpb   = (const float*)d_in[7];
  const float* bpb   = (const float*)d_in[8];
  const float* gamma = (const float*)d_in[9];
  const float* beta  = (const float*)d_in[10];
  float* out = (float*)d_out;

  const size_t qkvElems = (size_t)S_ * H_ * N_ * DH_;   // 8,388,608
  __bf16* qbf = (__bf16*)d_ws;
  __bf16* kbf = qbf + qkvElems;
  __bf16* vbf = kbf + qkvElems;
  __bf16* obf = vbf + qkvElems;
  float*  pb  = (float*)(obf + qkvElems);               // H*N*N f32

  k_qkv<<<dim3(512, 12), 256, 0, stream>>>(msa, Wqkv, bqkv, qbf, kbf, vbf);
  k_pairbias<<<dim3(2048), 256, 0, stream>>>(pair, Wpb, bpb, pb);
  k_attn<<<dim3(4096), 256, 98304, stream>>>(qbf, kbf, vbf, pb, mask, obf);
  k_outln<<<dim3(1024), 256, 0, stream>>>(obf, Wout, bout, msa, gamma, beta, out);
}
